// EquivariantSpatialConv_51926154609208
// MI455X (gfx1250) — matrix operand order repaired and checked by reading, not compiled, not run on hardware
//
#include <hip/hip_runtime.h>
#include <hip/hip_bf16.h>

// ---------------------------------------------------------------------------
// EquivariantSpatialConv for MI455X (gfx1250, wave32, WMMA bf16 split path)
//
//   out[n,k] = sum_{i,j} feat[n,i] * nb[n,j] * T[i,j,k]   (+ residual)
// == GEMM  Z[N x 484] @ T2[484 x 22],  Z[n, i*22+j] = feat[n,i]*nb[n,j].
//
// fp32 accuracy via split-bf16: x = hi + lo (two bf16);
//   a*b ~= a_hi*b_hi + a_hi*b_lo + a_lo*b_hi   (3x V_WMMA_F32_16X16X32_BF16)
// K padded 484 -> 512 = 16 chunks of 32.
//
// Kernel 1 builds T (Racah formula, fp64) and pre-swizzles bf16 hi/lo B
// panels into the exact WMMA per-lane layout. Kernel 2: 3x3 neighbor conv +
// WMMA GEMM + residual; B panel staged to LDS with async-to-LDS copies.
// ---------------------------------------------------------------------------

typedef __attribute__((ext_vector_type(16))) __bf16 v16bf;
typedef __attribute__((ext_vector_type(8)))  float  v8f;

#define NCH 22
#define KD 484
#define NCHUNK 16
#define ALPHA 0.5
// Bsw layout: [prec(2)][chunk(16)][half(2)][col(22)][ksub(16)]  (bf16)
#define BSW_ELEMS (2 * NCHUNK * 2 * NCH * 16)   // 22528 bf16 = 45056 B
#define ZSLOT BSW_ELEMS                          // 16-elem zero slot
#define T2F_WS_OFF 46080                         // byte offset of fp32 T in ws

__device__ __constant__ double FACT[16] = {
    1.0, 1.0, 2.0, 6.0, 24.0, 120.0, 720.0, 5040.0, 40320.0, 362880.0,
    3628800.0, 39916800.0, 479001600.0, 6227020800.0, 87178291200.0,
    1307674368000.0};

__device__ __constant__ int LS_[2]   = {4, 6};
__device__ __constant__ int OFFS_[2] = {0, 9};

struct C2 { double re, im; };
__device__ inline C2 cmul(C2 a, C2 b) {
    return {a.re * b.re - a.im * b.im, a.re * b.im + a.im * b.re};
}

__device__ double su2cg(int j1, int j2, int j3, int m1, int m2, int m3) {
    if (m1 + m2 != m3) return 0.0;
    if (m3 < -j3 || m3 > j3) return 0.0;
    double pref0 = (2.0 * j3 + 1.0) * FACT[j1 + j2 - j3] * FACT[j1 - j2 + j3] *
                   FACT[-j1 + j2 + j3] / FACT[j1 + j2 + j3 + 1];
    double pref = sqrt(pref0 * FACT[j3 + m3] * FACT[j3 - m3] * FACT[j1 - m1] *
                       FACT[j1 + m1] * FACT[j2 - m2] * FACT[j2 + m2]);
    int kmin = 0;
    if (-(j3 - j2 + m1) > kmin) kmin = -(j3 - j2 + m1);
    if (-(j3 - j1 - m2) > kmin) kmin = -(j3 - j1 - m2);
    int kmax = j1 + j2 - j3;
    if (j1 - m1 < kmax) kmax = j1 - m1;
    if (j2 + m2 < kmax) kmax = j2 + m2;
    double s = 0.0;
    for (int k = kmin; k <= kmax; ++k) {
        double d = FACT[k] * FACT[j1 + j2 - j3 - k] * FACT[j1 - m1 - k] *
                   FACT[j2 + m2 - k] * FACT[j3 - j2 + m1 + k] *
                   FACT[j3 - j1 - m2 + k];
        s += ((k & 1) ? -1.0 : 1.0) / d;
    }
    return pref * s;
}

// Nonzero entries of column `col` of q(l); includes (-i)^l phase (real here).
__device__ int q_col(int l, int col, int* rows, C2* vals) {
    double ph = (l == 6) ? -1.0 : 1.0;
    const double s2 = 0.70710678118654752440;
    int mu = col - l;
    if (mu == 0) { rows[0] = l; vals[0] = {ph, 0.0}; return 1; }
    if (mu > 0) {
        rows[0] = l - mu; vals[0] = {ph * s2, 0.0};
        rows[1] = l + mu; vals[1] = {ph * ((mu & 1) ? -s2 : s2), 0.0};
        return 2;
    }
    int nu = -mu;
    rows[0] = l - nu; vals[0] = {0.0, -ph * s2};
    rows[1] = l + nu; vals[1] = {0.0, ph * ((nu & 1) ? -s2 : s2)};
    return 2;
}

// Nonzero entries of row `row` of q(l) (for the conj(q3) factor).
__device__ int q_row(int l, int row, int* cols, C2* vals) {
    double ph = (l == 6) ? -1.0 : 1.0;
    const double s2 = 0.70710678118654752440;
    int m = row - l;
    if (m == 0) { cols[0] = l; vals[0] = {ph, 0.0}; return 1; }
    if (m < 0) {
        cols[0] = l - m; vals[0] = {ph * s2, 0.0};
        cols[1] = l + m; vals[1] = {0.0, -ph * s2};
        return 2;
    }
    cols[0] = l + m; vals[0] = {ph * ((m & 1) ? -s2 : s2), 0.0};
    cols[1] = l - m; vals[1] = {0.0, ph * ((m & 1) ? -s2 : s2)};
    return 2;
}

// ---------------------------------------------------------------------------
// Kernel 1: build fp32 T[484][22], then pre-swizzled bf16 hi/lo B panel.
// ---------------------------------------------------------------------------
__global__ __launch_bounds__(256) void build_T_kernel(const float* __restrict__ w_paths,
                                                      __bf16* __restrict__ BswG,
                                                      float* __restrict__ T2f) {
    __shared__ double CrS[13 * 13 * 13];
    __shared__ double red[256];
    int tid = threadIdx.x;

    for (int i = tid; i < KD * NCH; i += 256) T2f[i] = 0.0f;
    __syncthreads();

    for (int p = 0; p < 8; ++p) {
        int a = (p >> 2) & 1, b = (p >> 1) & 1, c = p & 1;
        int l1 = LS_[a], l2 = LS_[b], l3 = LS_[c];
        int d1 = 2 * l1 + 1, d2 = 2 * l2 + 1, d3 = 2 * l3 + 1;
        int tot = d1 * d2 * d3;

        double psum = 0.0;
        for (int e = tid; e < tot; e += 256) {
            int m = e % d3;
            int t = e / d3;
            int l = t % d2;
            int j = t / d2;
            int r1[2], r2[2], c3[2];
            C2 v1[2], v2[2], v3[2];
            int n1 = q_col(l1, j, r1, v1);
            int n2 = q_col(l2, l, r2, v2);
            int n3 = q_row(l3, m, c3, v3);
            C2 s = {0.0, 0.0};
            for (int ia = 0; ia < n1; ++ia)
                for (int ib = 0; ib < n2; ++ib) {
                    C2 q12 = cmul(v1[ia], v2[ib]);
                    int m1 = r1[ia] - l1, m2 = r2[ib] - l2;
                    for (int ic = 0; ic < n3; ++ic) {
                        int m3 = c3[ic] - l3;
                        double cg = su2cg(l1, l2, l3, m1, m2, m3);
                        if (cg != 0.0) {
                            C2 q3c = {v3[ic].re, -v3[ic].im};
                            C2 tt = cmul(q12, q3c);
                            s.re += tt.re * cg;
                            s.im += tt.im * cg;
                        }
                    }
                }
            CrS[e] = s.re;
            psum += s.re * s.re;
        }

        red[tid] = psum;
        __syncthreads();
        for (int step = 128; step > 0; step >>= 1) {
            if (tid < step) red[tid] += red[tid + step];
            __syncthreads();
        }
        double norm = sqrt(red[0]);
        double scale = ALPHA * (double)w_paths[p] / norm;
        __syncthreads();

        for (int e = tid; e < tot; e += 256) {
            int m = e % d3;
            int t = e / d3;
            int l = t % d2;
            int j = t / d2;
            int row = (OFFS_[a] + j) * NCH + (OFFS_[b] + l);  // ij index (K)
            int col = OFFS_[c] + m;                           // output k (N)
            T2f[row * NCH + col] = (float)(scale * CrS[e]);
        }
        __syncthreads();
    }

    // ---- Phase 2: swizzle into bf16 hi/lo WMMA B layout ----
    // flat = (((prec*16 + chunk)*2 + h)*22 + col)*16 + ksub
    for (int ei = tid; ei < BSW_ELEMS; ei += 256) {
        int ksub = ei & 15;
        int t = ei >> 4;
        int col = t % NCH;
        int u = t / NCH;
        int h = u & 1;
        int ch = (u >> 1) & 15;
        int prec = u >> 5;
        int kg = 32 * ch + 16 * h + ksub;
        float f = (kg < KD) ? T2f[kg * NCH + col] : 0.0f;
        __bf16 hi = (__bf16)f;
        __bf16 v = (prec == 0) ? hi : (__bf16)(f - (float)hi);
        BswG[ei] = v;
    }
}

// ---------------------------------------------------------------------------
// Kernel 2: 3x3 neighbor average + split-bf16 WMMA GEMM + residual.
// 128 threads = 4 waves; each wave owns 16 pixels (one WMMA M-tile).
// ---------------------------------------------------------------------------
__global__ __launch_bounds__(128) void tp_wmma_kernel(
    const float* __restrict__ f4, const float* __restrict__ f6,
    const float* __restrict__ sw, const __bf16* __restrict__ BswG,
    const int* __restrict__ Hp, const int* __restrict__ Wp,
    float* __restrict__ out, int N) {
    __shared__ __bf16 Bsw[BSW_ELEMS + 16];  // +16 bf16 zero slot = 45,088 B
    __shared__ float Flds[64 * NCH];        // 5,632 B (own features)
    __shared__ float NBlds[64 * NCH];       // 5,632 B (neighbor average)

    int tid = threadIdx.x;
    int wave = tid >> 5;
    int lane = tid & 31;
    int half = lane >> 4;
    int pix = lane & 15;

    // Stage B panel global -> LDS via CDNA5 async-to-LDS (ASYNCcnt tracked).
    {
        unsigned ldsBase = (unsigned)(size_t)(&Bsw[0]);
        // 45,056 B / 16 B = 2816 b128 transfers; 128 threads -> 22 iters.
        for (int i = tid; i < BSW_ELEMS / 8; i += 128) {
            unsigned ldsAddr = ldsBase + (unsigned)i * 16u;
            unsigned long long gaddr = (unsigned long long)(size_t)(BswG) +
                                       (unsigned long long)i * 16ull;
            asm volatile("global_load_async_to_lds_b128 %0, %1, off"
                         :
                         : "v"(ldsAddr), "v"(gaddr)
                         : "memory");
        }
        if (tid < 8) ((unsigned*)(&Bsw[ZSLOT]))[tid] = 0u;  // zero slot
    }

    int H = *Hp, W = *Wp;
    int pl = wave * 16 + pix;      // pixel slot within workgroup
    int n = blockIdx.x * 64 + pl;  // global pixel id
    int nl = n < N ? n : (N - 1);  // clamp loads; EXEC stays full
    int x = nl % W;
    int y = nl / W;

    float swv[9];
#pragma unroll
    for (int t = 0; t < 9; ++t) swv[t] = sw[t];

    // Two lanes per pixel: each computes 11 of 22 channels (own + neighbor).
    int chBase = half * 11;
#pragma unroll
    for (int cc = 0; cc < 11; ++cc) {
        int ch = chBase + cc;
        float fv = (ch < 9) ? f4[nl * 9 + ch] : f6[nl * 13 + (ch - 9)];
        float acc = 0.0f;
#pragma unroll
        for (int dy = -1; dy <= 1; ++dy) {
#pragma unroll
            for (int dx = -1; dx <= 1; ++dx) {
                int yy = y + dy; yy = yy < 0 ? 0 : (yy >= H ? H - 1 : yy);
                int xx = x + dx; xx = xx < 0 ? 0 : (xx >= W ? W - 1 : xx);
                int nn = yy * W + xx;
                float v = (ch < 9) ? f4[nn * 9 + ch] : f6[nn * 13 + (ch - 9)];
                acc += swv[(dy + 1) * 3 + (dx + 1)] * v;
            }
        }
        Flds[pl * NCH + ch] = fv;
        NBlds[pl * NCH + ch] = acc;
    }

    asm volatile("s_wait_asynccnt 0" ::: "memory");
    __syncthreads();

    // GEMM: D[16 x 32] = Z[16 x 512] * B[512 x 32], split-bf16 (3 products).
    v8f acc0 = {};  // output cols 0..15
    v8f acc1 = {};  // output cols 16..21
    const float* Fp = &Flds[pl * NCH];
    const float* Np = &NBlds[pl * NCH];
    int col0 = pix;
    int col1 = 16 + pix;
    bool c1ok = (col1 < NCH);

#pragma unroll
    for (int c = 0; c < NCHUNK; ++c) {
        // ---- Build A hi/lo (z products). K indices are compile-time pairs
        //      selected by `half` (half==1 lane covers K = k0 + 16). ----
        v16bf ahi, alo;
#pragma unroll
        for (int e = 0; e < 16; ++e) {
            const int k0 = 32 * c + e;  // half==0 K slot
            const int k1 = k0 + 16;     // half==1 K slot
            float z;
            if (k1 < KD) {
                int ii = half ? (k1 / NCH) : (k0 / NCH);
                int jj = half ? (k1 % NCH) : (k0 % NCH);
                z = Fp[ii] * Np[jj];
            } else if (k0 < KD) {
                float zz = Fp[k0 / NCH] * Np[k0 % NCH];
                z = half ? 0.0f : zz;
            } else {
                z = 0.0f;
            }
            __bf16 h = (__bf16)z;
            ahi[e] = h;
            alo[e] = (__bf16)(z - (float)h);
        }

        // ---- B operands: 32 contiguous bytes per lane (2x ds_load_b128);
        //      out-of-range tile-1 columns redirect to the LDS zero slot. ----
        int oh0 = (((0 * NCHUNK + c) * 2 + half) * NCH + col0) * 16;
        int ol0 = (((1 * NCHUNK + c) * 2 + half) * NCH + col0) * 16;
        int oh1 = c1ok ? (((0 * NCHUNK + c) * 2 + half) * NCH + col1) * 16 : ZSLOT;
        int ol1 = c1ok ? (((1 * NCHUNK + c) * 2 + half) * NCH + col1) * 16 : ZSLOT;
        v16bf bh0 = *(const v16bf*)(Bsw + oh0);
        v16bf bl0 = *(const v16bf*)(Bsw + ol0);
        v16bf bh1 = *(const v16bf*)(Bsw + oh1);
        v16bf bl1 = *(const v16bf*)(Bsw + ol1);

        acc0 = __builtin_amdgcn_wmma_f32_16x16x32_bf16(false, ahi, false, bh0,
                                                       (short)0, acc0, false, false);
        acc0 = __builtin_amdgcn_wmma_f32_16x16x32_bf16(false, ahi, false, bl0,
                                                       (short)0, acc0, false, false);
        acc0 = __builtin_amdgcn_wmma_f32_16x16x32_bf16(false, alo, false, bh0,
                                                       (short)0, acc0, false, false);
        acc1 = __builtin_amdgcn_wmma_f32_16x16x32_bf16(false, ahi, false, bh1,
                                                       (short)0, acc1, false, false);
        acc1 = __builtin_amdgcn_wmma_f32_16x16x32_bf16(false, ahi, false, bl1,
                                                       (short)0, acc1, false, false);
        acc1 = __builtin_amdgcn_wmma_f32_16x16x32_bf16(false, alo, false, bh1,
                                                       (short)0, acc1, false, false);
    }

    // Scatter D + residual.  D layout: VGPR r -> M = r + 8*half, N = lane&15.
#pragma unroll
    for (int r = 0; r < 8; ++r) {
        int M = r + half * 8;
        int n_out = blockIdx.x * 64 + wave * 16 + M;
        if (n_out < N) {
            int k0 = pix;
            float v0 = acc0[r];
            if (k0 < 9) {
                out[n_out * 9 + k0] = v0 + f4[n_out * 9 + k0];
            } else {
                out[(size_t)N * 9 + n_out * 13 + (k0 - 9)] =
                    v0 + f6[n_out * 13 + (k0 - 9)];
            }
            int k1 = 16 + pix;
            if (k1 < NCH) {
                float v1 = acc1[r];
                out[(size_t)N * 9 + n_out * 13 + (k1 - 9)] =
                    v1 + f6[n_out * 13 + (k1 - 9)];
            }
        }
    }
}

// ---------------------------------------------------------------------------
extern "C" void kernel_launch(void* const* d_in, const int* in_sizes, int n_in,
                              void* d_out, int out_size, void* d_ws, size_t ws_size,
                              hipStream_t stream) {
    const float* f4 = (const float*)d_in[0];
    const float* f6 = (const float*)d_in[1];
    const float* sw = (const float*)d_in[2];
    const float* wp = (const float*)d_in[3];
    const int* Hp = (const int*)d_in[4];
    const int* Wp = (const int*)d_in[5];
    float* out = (float*)d_out;

    __bf16* BswG = (__bf16*)d_ws;                          // 45,056 B
    float* T2f = (float*)((char*)d_ws + T2F_WS_OFF);       // 42,592 B

    int N = in_sizes[0] / 9;

    build_T_kernel<<<1, 256, 0, stream>>>(wp, BswG, T2f);

    int blocks = (N + 63) / 64;
    tp_wmma_kernel<<<blocks, 128, 0, stream>>>(f4, f6, sw, BswG, Hp, Wp, out, N);
}
